// AttentionGraphModel_28355374088754
// MI455X (gfx1250) — matrix operand order repaired and checked by reading, not compiled, not run on hardware
//
#include <hip/hip_runtime.h>
#include <hip/hip_bf16.h>

#define LEAK 0.2f

typedef __attribute__((ext_vector_type(16))) __bf16 v16bf;
typedef __attribute__((ext_vector_type(8)))  __bf16 v8bf;
typedef __attribute__((ext_vector_type(8)))  float  v8f;
typedef __attribute__((ext_vector_type(4)))  float  vf4;

static __device__ inline v16bf cat8(v8bf lo, v8bf hi) {
  v16bf r;
#pragma unroll
  for (int i = 0; i < 8; ++i) { r[i] = lo[i]; r[i + 8] = hi[i]; }
  return r;
}

static __device__ inline v8bf cvt8(vf4 a, vf4 b) {
  v8bf r;
#pragma unroll
  for (int i = 0; i < 4; ++i) { r[i] = (__bf16)a[i]; r[i + 4] = (__bf16)b[i]; }
  return r;
}

// ---------------------------------------------------------------------------
// Pack dense fp32 adjacency into a bitmask: 256 MB streamed ONCE (nontemporal
// b128 loads), 8 MB written. Subsequent layers read only the bitmask.
// ---------------------------------------------------------------------------
__global__ void pack_adj(const float* __restrict__ adj,
                         unsigned int* __restrict__ mask, int N) {
  size_t w = (size_t)blockIdx.x * blockDim.x + threadIdx.x;
  size_t total = ((size_t)N * (size_t)N) >> 5;
  if (w >= total) return;
  const vf4* p = (const vf4*)(adj + (w << 5));
  unsigned int bits = 0u;
#pragma unroll
  for (int q = 0; q < 8; ++q) {
    vf4 v = __builtin_nontemporal_load(p + q);
#pragma unroll
    for (int i = 0; i < 4; ++i)
      bits |= (v[i] != 0.0f ? 1u : 0u) << (q * 4 + i);
  }
  mask[w] = bits;
}

// ---------------------------------------------------------------------------
// h = leaky_relu(x @ W^T + b), output bf16. One 16-row i-tile per block,
// wave g owns 16 output features. A/B fragments loaded directly from global
// fp32 in the ISA 16-bit-operand VGPR striping, converted to bf16 in-reg.
// ---------------------------------------------------------------------------
template <int F>
__global__ __launch_bounds__(32 * (F / 16)) void gemm_lrelu(
    const float* __restrict__ x, const float* __restrict__ W,
    const float* __restrict__ b, __bf16* __restrict__ hout, int N) {
  constexpr int D = 128;
  const int lane = threadIdx.x & 31;
  const int g    = threadIdx.x >> 5;
  const int i0   = blockIdx.x * 16;
  const int mn   = lane & 15;   // A row / B column within tile
  const int hf   = lane >> 4;   // lane half selects K sub-block
  const int fb   = g * 16;

  v8f acc = {};
  const float* xrow = x + (size_t)(i0 + mn) * D;  // A: row m of x
  const float* wrow = W + (size_t)(fb + mn) * D;  // B: (K,n) = W[f][d]

#pragma unroll
  for (int kk = 0; kk < D; kk += 32) {
    // A 16x32: lane<16 holds K{0..7,16..23}, lane>=16 K{8..15,24..31}
    const int ka = kk + hf * 8;
    vf4 a0 = *(const vf4*)(xrow + ka);
    vf4 a1 = *(const vf4*)(xrow + ka + 4);
    vf4 a2 = *(const vf4*)(xrow + ka + 16);
    vf4 a3 = *(const vf4*)(xrow + ka + 20);
    v16bf A = cat8(cvt8(a0, a1), cvt8(a2, a3));
    // B 32x16: lane<16 holds K{0..7,16..23}, lane>=16 K{8..15,24..31}
    const int kb = kk + hf * 8;
    vf4 b0 = *(const vf4*)(wrow + kb);
    vf4 b1 = *(const vf4*)(wrow + kb + 4);
    vf4 b2 = *(const vf4*)(wrow + kb + 16);
    vf4 b3 = *(const vf4*)(wrow + kb + 20);
    v16bf B = cat8(cvt8(b0, b1), cvt8(b2, b3));
    acc = __builtin_amdgcn_wmma_f32_16x16x32_bf16(false, A, false, B,
                                                  (short)0, acc, false, false);
  }

  const int fcol = fb + mn;
  const float bias = b[fcol];
#pragma unroll
  for (int v = 0; v < 8; ++v) {
    const int row = i0 + v + hf * 8;  // C layout: VGPR v -> M=v / M=8+v
    float val = acc[v] + bias;
    val = val > 0.0f ? val : LEAK * val;
    hout[(size_t)row * F + fcol] = (__bf16)val;
  }
}

// ---------------------------------------------------------------------------
// el[i] = h[i,:] . aW[:F] + ab ; er[i] = h[i,:] . aW[F:]
// ---------------------------------------------------------------------------
template <int F>
__global__ void attn_el_er(const __bf16* __restrict__ h,
                           const float* __restrict__ aW,
                           const float* __restrict__ ab,
                           float* __restrict__ el, float* __restrict__ er) {
  __shared__ float sl[F];
  __shared__ float sr[F];
  const int r = blockIdx.x, f = threadIdx.x;
  const float hv = (float)h[(size_t)r * F + f];
  sl[f] = hv * aW[f];
  sr[f] = hv * aW[F + f];
  __syncthreads();
  for (int s = F / 2; s > 0; s >>= 1) {
    if (f < s) { sl[f] += sl[f + s]; sr[f] += sr[f + s]; }
    __syncthreads();
  }
  if (f == 0) { el[r] = sl[0] + ab[0]; er[r] = sr[0]; }
}

// ---------------------------------------------------------------------------
// Fused masked-softmax attention aggregation:
//   out[i,:] = relu( (sum_j w_ij * h[j,:]) / (sum_j w_ij) )
//   w_ij = adjbit(i,j) * exp(el[i] + er[j])        (ab folded into el)
// 32-row i-tile per block; 16 waves (F=128): two 8-wave row-groups share one
// staged 32xF h tile, halving L2->LDS h traffic vs a 16-row tile. One bf16
// WMMA per wave per 32-wide j tile.
// ---------------------------------------------------------------------------
template <int F>
__global__ __launch_bounds__(64 * (F / 16)) void agg_softmax(
    const unsigned int* __restrict__ mask, const __bf16* __restrict__ h,
    const float* __restrict__ el, const float* __restrict__ er,
    float* __restrict__ out, int N) {
  constexpr int WPF  = F / 16;      // waves per f-slice group (8 / 4)
  constexpr int NW   = WPF * 2;     // waves per block (two row-groups)
  constexpr int NT   = NW * 32;     // threads per block (512 / 256)
  constexpr int ROWS = 32;          // i-rows per block
  constexpr int EPTW = ROWS * 32 / NT;     // w elements per thread (2 / 4)
  constexpr int EPTH = 16 * F / NT;        // hT b32-pairs per thread (4 / 4)
  constexpr int LDW  = 40;          // LDS row stride (80 B: 16B-aligned,
                                    // bank stride 20 -> conflict-free frags)
  __shared__ __align__(16) __bf16 w_lds[ROWS * LDW];
  __shared__ __align__(16) __bf16 hT[F * LDW];
  __shared__ float el_s[ROWS];
  __shared__ float den_s[ROWS];

  const int tid  = threadIdx.x;
  const int lane = tid & 31;
  const int g    = tid >> 5;
  const int i0   = blockIdx.x * ROWS;
  const unsigned short* h16 = (const unsigned short*)h;

  if (tid < ROWS) el_s[tid] = el[i0 + tid];

  float dsum[EPTW];
#pragma unroll
  for (int e = 0; e < EPTW; ++e) dsum[e] = 0.0f;
  v8f acc = {};

  const int mn  = lane & 15;
  const int hf  = lane >> 4;
  const int g_f = g % WPF;          // f-slice of this wave
  const int g_r = g / WPF;          // row-group (0 or 1)

  for (int j0 = 0; j0 < N; j0 += 32) {
    __syncthreads();  // LDS reuse fence (also covers el_s on first iter)
    // --- stage w tile: m = g + e*NW is wave-uniform, k = lane ---
#pragma unroll
    for (int e = 0; e < EPTW; ++e) {
      const int m = g + e * NW;
      const unsigned int bits = mask[((size_t)(i0 + m) * N + j0) >> 5];
      float wv = 0.0f;
      if ((bits >> lane) & 1u) wv = __expf(el_s[m] + er[j0 + lane]);
      w_lds[m * LDW + lane] = (__bf16)wv;
      dsum[e] += wv;
    }
    // --- stage h tile transposed, two j per b32 store: hT[f][j] = h[j0+j][f]
#pragma unroll
    for (int e = 0; e < EPTH; ++e) {
      const int idx = tid + e * NT;
      const int f   = idx % F;
      const int j   = (idx / F) * 2;
      const unsigned int lo = h16[(size_t)(j0 + j) * F + f];
      const unsigned int hi = h16[(size_t)(j0 + j + 1) * F + f];
      *(unsigned int*)(&hT[f * LDW + j]) = lo | (hi << 16);
    }
    __syncthreads();
    // --- fragments + WMMA ---
    const __bf16* ap = &w_lds[(g_r * 16 + mn) * LDW + hf * 8];
    v16bf A = cat8(*(const v8bf*)ap, *(const v8bf*)(ap + 16));
    const __bf16* bp = &hT[(g_f * 16 + mn) * LDW + hf * 8];
    v16bf B = cat8(*(const v8bf*)bp, *(const v8bf*)(bp + 16));
    acc = __builtin_amdgcn_wmma_f32_16x16x32_bf16(false, A, false, B,
                                                  (short)0, acc, false, false);
  }

  // --- denominator: reduce each wave's 32 lanes (one row per (g,e)) ---
#pragma unroll
  for (int e = 0; e < EPTW; ++e) {
    float v = dsum[e];
#pragma unroll
    for (int off = 16; off; off >>= 1) v += __shfl_xor(v, off, 32);
    if (lane == 0) den_s[g + e * NW] = v;
  }
  __syncthreads();

  const int fcol = g_f * 16 + mn;
#pragma unroll
  for (int v = 0; v < 8; ++v) {
    const int row = g_r * 16 + v + hf * 8;
    float val = acc[v] / den_s[row];
    val = val > 0.0f ? val : 0.0f;
    out[(size_t)(i0 + row) * F + fcol] = val;
  }
}

// ---------------------------------------------------------------------------
extern "C" void kernel_launch(void* const* d_in, const int* in_sizes, int n_in,
                              void* d_out, int out_size, void* d_ws,
                              size_t ws_size, hipStream_t stream) {
  (void)in_sizes; (void)n_in; (void)out_size; (void)ws_size;
  const int N = 8192, Dh = 128;

  const float* x   = (const float*)d_in[0];
  const float* adj = (const float*)d_in[1];
  const float* W0  = (const float*)d_in[2];
  const float* b0  = (const float*)d_in[3];
  const float* aW0 = (const float*)d_in[4];
  const float* ab0 = (const float*)d_in[5];
  const float* W1  = (const float*)d_in[6];
  const float* b1  = (const float*)d_in[7];
  const float* aW1 = (const float*)d_in[8];
  const float* ab1 = (const float*)d_in[9];
  const float* W2  = (const float*)d_in[10];
  const float* b2  = (const float*)d_in[11];
  const float* aW2 = (const float*)d_in[12];
  const float* ab2 = (const float*)d_in[13];

  char* ws = (char*)d_ws;
  size_t off = 0;
  auto carve = [&](size_t bytes) {
    void* p = ws + off;
    off += (bytes + 255) & ~(size_t)255;
    return p;
  };
  unsigned int* mask = (unsigned int*)carve((size_t)N * N / 8);  // 8 MB
  __bf16* hbf = (__bf16*)carve((size_t)N * Dh * sizeof(__bf16)); // 2 MB
  float*  el  = (float*)carve((size_t)N * sizeof(float));
  float*  er  = (float*)carve((size_t)N * sizeof(float));
  float*  xA  = (float*)carve((size_t)N * Dh * sizeof(float));   // 4 MB
  float*  xB  = (float*)carve((size_t)N * Dh * sizeof(float));   // 4 MB

  const int maskWords = (int)(((size_t)N * N) >> 5);
  pack_adj<<<(maskWords + 255) / 256, 256, 0, stream>>>(adj, mask, N);

  // layer 0: 128 -> 128
  gemm_lrelu<128><<<N / 16, 256, 0, stream>>>(x, W0, b0, hbf, N);
  attn_el_er<128><<<N, 128, 0, stream>>>(hbf, aW0, ab0, el, er);
  agg_softmax<128><<<N / 32, 512, 0, stream>>>(mask, hbf, el, er, xA, N);
  // layer 1: 128 -> 128
  gemm_lrelu<128><<<N / 16, 256, 0, stream>>>(xA, W1, b1, hbf, N);
  attn_el_er<128><<<N, 128, 0, stream>>>(hbf, aW1, ab1, el, er);
  agg_softmax<128><<<N / 32, 512, 0, stream>>>(mask, hbf, el, er, xB, N);
  // layer 2: 128 -> 64
  gemm_lrelu<64><<<N / 16, 128, 0, stream>>>(xB, W2, b2, hbf, N);
  attn_el_er<64><<<N, 64, 0, stream>>>(hbf, aW2, ab2, el, er);
  agg_softmax<64><<<N / 32, 256, 0, stream>>>(mask, hbf, el, er,
                                              (float*)d_out, N);
}
